// TransformerDeepRV_39874476376491
// MI455X (gfx1250) — matrix operand order repaired and checked by reading, not compiled, not run on hardware
//
#include <hip/hip_runtime.h>
#include <cstdint>
#include <cstddef>

// ---------------------------------------------------------------------------
// TransformerDeepRV forward for MI455X (gfx1250, wave32, WMMA).
// All heavy GEMMs run as v_wmma_f32_16x16x32_f16 (f16 in, f32 accum).
// Residual stream kept in f32; weights converted to f16 once per launch.
// Input pointer order assumed = setup_inputs() dict insertion order:
//   0:z 1:ls 2:s 3:embed 4:in_W1 5:in_b1 6:in_W2 7:in_b2 8:rbf_W 9:rbf_b
//   10:out_W1 11:out_b1 12:out_W2 13:out_b2 14:out_W3 15:out_b3
//   16+16*blk: ln1_g ln1_b Wq bq Wk bk Wv bv Wo bo ln2_g ln2_b Wf1 bf1 Wf2 bf2
// ---------------------------------------------------------------------------

#define DMODEL 128
#define LSEQ   1024
#define BATCH  16
#define QKVW   256   // 2*D
#define HDIM   64

typedef __attribute__((ext_vector_type(16))) _Float16 v16h;
typedef __attribute__((ext_vector_type(8)))  _Float16 v8h;
typedef __attribute__((ext_vector_type(8)))  float    v8f;

__device__ __forceinline__ v8f wmma16(v16h a, v16h b, v8f c) {
  // v_wmma_f32_16x16x32_f16  (neg_a, A, neg_b, B, c_mod, C, reuse_a, reuse_b)
  return __builtin_amdgcn_wmma_f32_16x16x32_f16(false, a, false, b, (short)0, c,
                                                false, false);
}

// A-fragment: halves [0..7] from p[0..7], halves [8..15] from p[8..15]
// (per-lane K-half selection is already folded into p by the caller).
__device__ __forceinline__ v16h ldfragA(const _Float16* p) {
  v8h lo = *(const v8h*)p;
  v8h hi = *(const v8h*)(p + 8);
  return __builtin_shufflevector(lo, hi, 0, 1, 2, 3, 4, 5, 6, 7,
                                 8, 9, 10, 11, 12, 13, 14, 15);
}

__device__ __forceinline__ float gelu_f(float x) {
  float x3 = x * x * x;
  return 0.5f * x * (1.0f + tanhf(0.7978845608f * (x + 0.044715f * x3)));
}

// Reductions across a 16-lane half of a wave32 (xor masks <= 8 stay in-half).
__device__ __forceinline__ float hmax16(float v) {
  v = fmaxf(v, __shfl_xor(v, 1, 32));
  v = fmaxf(v, __shfl_xor(v, 2, 32));
  v = fmaxf(v, __shfl_xor(v, 4, 32));
  v = fmaxf(v, __shfl_xor(v, 8, 32));
  return v;
}
__device__ __forceinline__ float hsum16(float v) {
  v += __shfl_xor(v, 1, 32);
  v += __shfl_xor(v, 2, 32);
  v += __shfl_xor(v, 4, 32);
  v += __shfl_xor(v, 8, 32);
  return v;
}

// ---------------------------------------------------------------------------
// Elementwise f32 -> f16
__global__ void k_f32_to_f16(const float* __restrict__ in, _Float16* __restrict__ out, int n) {
  int i = blockIdx.x * 256 + threadIdx.x;
  if (i < n) out[i] = (_Float16)in[i];
}

// ---------------------------------------------------------------------------
// Featurize: X[b,l,:] = [embed(l, 0..253) | z(b,l) | ls]  -> f16 (16384 x 256)
__global__ void k_featurize(const float* __restrict__ emb, const float* __restrict__ z,
                            const float* __restrict__ ls, _Float16* __restrict__ X) {
  int row = blockIdx.x, c = threadIdx.x;
  int b = row >> 10, l = row & 1023;
  float v;
  if (c < 254)       v = emb[l * 254 + c];
  else if (c == 254) v = z[b * 1024 + l];
  else               v = ls[0];
  X[(size_t)row * 256 + c] = (_Float16)v;
}

// ---------------------------------------------------------------------------
// LayerNorm over D=128, one row per 128-thread block; f16 output.
__global__ void k_layernorm(const float* __restrict__ X, const float* __restrict__ g,
                            const float* __restrict__ bt, _Float16* __restrict__ out) {
  __shared__ float red[128];
  int row = blockIdx.x, tid = threadIdx.x;
  float x = X[(size_t)row * DMODEL + tid];
  red[tid] = x; __syncthreads();
  for (int s2 = 64; s2 > 0; s2 >>= 1) { if (tid < s2) red[tid] += red[tid + s2]; __syncthreads(); }
  float mu = red[0] * (1.0f / DMODEL);
  __syncthreads();
  float d = x - mu;
  red[tid] = d * d; __syncthreads();
  for (int s2 = 64; s2 > 0; s2 >>= 1) { if (tid < s2) red[tid] += red[tid + s2]; __syncthreads(); }
  float var = red[0] * (1.0f / DMODEL);
  float y = d * rsqrtf(var + 1e-6f) * g[tid] + bt[tid];
  out[(size_t)row * DMODEL + tid] = (_Float16)y;
}

// ---------------------------------------------------------------------------
// RBF distance bias: bias[h][i][j], MU_k = 0.1k, GAMMA = 50.
__global__ void k_rbf(const float* __restrict__ s, const float* __restrict__ W,
                      const float* __restrict__ bvec, float* __restrict__ bias) {
  int idx = blockIdx.x * 256 + threadIdx.x;
  int i = idx >> 10, j = idx & 1023;
  float dx = s[2 * i] - s[2 * j], dy = s[2 * i + 1] - s[2 * j + 1];
  float d = sqrtf(dx * dx + dy * dy);
  float a0 = bvec[0], a1 = bvec[1], a2 = bvec[2], a3 = bvec[3];
#pragma unroll
  for (int k = 0; k < 16; ++k) {
    float t = d - 0.1f * (float)k;
    float f = __expf(-50.0f * t * t);
    a0 += f * W[k * 4 + 0]; a1 += f * W[k * 4 + 1];
    a2 += f * W[k * 4 + 2]; a3 += f * W[k * 4 + 3];
  }
  bias[0u * 1048576u + idx] = a0;
  bias[1u * 1048576u + idx] = a1;
  bias[2u * 1048576u + idx] = a2;
  bias[3u * 1048576u + idx] = a3;
}

// ---------------------------------------------------------------------------
// Generic WMMA GEMM: C(M,N) = act(A(M,K) @ B(K,N) + bias) [+ res]
// Block tile 64x128, K-step 32, 8 waves (4x2), wave tile 16x64 (4 wmma/K-step).
// A staged row-major (frag reads contiguous); B staged TRANSPOSED in LDS so
// each B fragment is one contiguous 32-byte LDS load.
template <bool GELU>
__global__ void k_gemm(const _Float16* __restrict__ A, const _Float16* __restrict__ Bw,
                       const float* __restrict__ bias, const float* __restrict__ res,
                       float* __restrict__ outF, _Float16* __restrict__ outH,
                       int M, int N, int K) {
  __shared__ _Float16 As[64 * 32];      // [row][k]
  __shared__ _Float16 Bst[128 * 32];    // transposed: [col][k]
  const int tid = threadIdx.x, lane = tid & 31, wid = tid >> 5;
  const int wm = wid >> 1, wn = wid & 1;
  const int rowBase = blockIdx.y * 64, colBase = blockIdx.x * 128;
  (void)M;
  v8f acc[4] = {};
  const int kb   = (lane >= 16) ? 8  : 0;  // unused; A-frag K-half select also uses kk0
  const int kk0  = (lane >= 16) ? 16 : 0;  // B-frag K-half select
  const int bcol = wn * 64 + (lane & 15);
  // staging indices (one 8-half vector for A, two for B, per thread)
  const int aIdx = tid * 8;                 // 2048 halves
  const int aR = aIdx >> 5, aC = aIdx & 31;
  for (int kt = 0; kt < K; kt += 32) {
    __syncthreads();
    *(v8h*)&As[aIdx] = *(const v8h*)&A[(size_t)(rowBase + aR) * K + kt + aC];
#pragma unroll
    for (int t = 0; t < 2; ++t) {           // B tile 32x128 -> Bst[128][32]
      int idx = (tid * 2 + t) * 8;
      int r = idx >> 7, c = idx & 127;
      v8h gl = *(const v8h*)&Bw[(size_t)(kt + r) * N + colBase + c];
#pragma unroll
      for (int e = 0; e < 8; ++e) Bst[(c + e) * 32 + r] = gl[e];
    }
    __syncthreads();
    v16h a = ldfragA(&As[(wm * 16 + (lane & 15)) * 32 + kk0]);
#pragma unroll
    for (int nt = 0; nt < 4; ++nt) {
      v16h bf = *(const v16h*)&Bst[(bcol + nt * 16) * 32 + kk0];
      acc[nt] = wmma16(a, bf, acc[nt]);
    }
  }
  const int roff = (lane >= 16) ? 8 : 0;
#pragma unroll
  for (int nt = 0; nt < 4; ++nt) {
    const int col = colBase + wn * 64 + nt * 16 + (lane & 15);
    const float bc = bias[col];
#pragma unroll
    for (int i = 0; i < 8; ++i) {
      const int row = rowBase + wm * 16 + roff + i;
      float v = acc[nt][i] + bc;
      if (GELU) v = gelu_f(v);
      size_t o = (size_t)row * N + col;
      if (res)  v += res[o];
      if (outF) outF[o] = v;
      if (outH) outH[o] = (_Float16)v;
    }
  }
}

// ---------------------------------------------------------------------------
// Flash attention with RBF bias. Block = 8 waves = 128 q rows for one (b,h).
// K tile staged row-major (contraction dim d contiguous -> contiguous frags);
// V tile staged TRANSPOSED (contraction dim = key row) so P@V B-frags are
// contiguous 32-byte LDS loads. Online softmax fused to minimize live ranges.
__global__ void k_attn(const _Float16* __restrict__ Q, const _Float16* __restrict__ Kt,
                       const _Float16* __restrict__ V, const float* __restrict__ bias,
                       _Float16* __restrict__ O) {
  __shared__ _Float16 Ks[32 * 64];      // [key][d]
  __shared__ _Float16 Vt[64 * 32];      // transposed: [d][key]
  __shared__ _Float16 Ps[8][16 * 32];   // per-wave P tile
  const int tid = threadIdx.x, lane = tid & 31, wid = tid >> 5;
  const int qb = blockIdx.x & 7, hd = (blockIdx.x >> 3) & 3, b = blockIdx.x >> 5;
  const int qbase = qb * 128 + wid * 16;
  const _Float16* Qp = Q  + (size_t)b * LSEQ * QKVW + hd * HDIM;
  const _Float16* Kp = Kt + (size_t)b * LSEQ * QKVW + hd * HDIM;
  const _Float16* Vp = V  + (size_t)b * LSEQ * QKVW + hd * HDIM;

  const int kb   = (lane >= 16) ? 8  : 0;
  const int kk0  = (lane >= 16) ? 16 : 0;
  const int roff = (lane >= 16) ? 8  : 0;
  const int n0   = lane & 15;

  // Per-lane bias base: rows qbase+roff+0..7, col = j + n0 (+16). Offsets i*LSEQ
  // are compile-time immediates after unrolling.
  const float* bRow = bias + (size_t)hd * LSEQ * LSEQ
                    + (size_t)(qbase + roff) * LSEQ + n0;

  // Q fragments: 16x64 -> two 16x32 A-frags, via 16-byte vector loads.
  const _Float16* qp = Qp + (size_t)(qbase + n0) * QKVW;
  v16h qa0 = ldfragA(qp + kk0);
  v16h qa1 = ldfragA(qp + 32 + kk0);

  v8f oac[4] = {};
  float m[8], lsum[8];
#pragma unroll
  for (int i = 0; i < 8; ++i) { m[i] = -1e30f; lsum[i] = 0.f; }

  // staging: each thread moves one 8-half vector of K and of V per tile
  const int sIdx = tid * 8;                 // 2048 halves per tile
  const int sR = sIdx >> 6, sC = sIdx & 63;

  for (int j = 0; j < LSEQ; j += 32) {
    __syncthreads();
    *(v8h*)&Ks[sIdx] = *(const v8h*)&Kp[(size_t)(j + sR) * QKVW + sC];
    v8h gv = *(const v8h*)&Vp[(size_t)(j + sR) * QKVW + sC];
#pragma unroll
    for (int e = 0; e < 8; ++e) Vt[(sC + e) * 32 + sR] = gv[e];
    __syncthreads();

    v8f s0 = {}, s1 = {};
    s0 = wmma16(qa0, *(const v16h*)&Ks[n0 * 64 + kk0],             s0);
    s0 = wmma16(qa1, *(const v16h*)&Ks[n0 * 64 + 32 + kk0],        s0);
    s1 = wmma16(qa0, *(const v16h*)&Ks[(16 + n0) * 64 + kk0],      s1);
    s1 = wmma16(qa1, *(const v16h*)&Ks[(16 + n0) * 64 + 32 + kk0], s1);

    _Float16* Pw = Ps[wid];
    const float* bj = bRow + j;
    float corr[8];
#pragma unroll
    for (int i = 0; i < 8; ++i) {
      float v0 = s0[i] * 0.125f + bj[i * LSEQ];
      float v1 = s1[i] * 0.125f + bj[i * LSEQ + 16];
      float rm = hmax16(fmaxf(v0, v1));
      float mn = fmaxf(m[i], rm);
      corr[i] = __expf(m[i] - mn);
      m[i] = mn;
      float p0 = __expf(v0 - mn);
      float p1 = __expf(v1 - mn);
      lsum[i] = lsum[i] * corr[i] + hsum16(p0 + p1);
      Pw[(roff + i) * 32 + n0]      = (_Float16)p0;
      Pw[(roff + i) * 32 + 16 + n0] = (_Float16)p1;
    }
#pragma unroll
    for (int nt = 0; nt < 4; ++nt)
#pragma unroll
      for (int i = 0; i < 8; ++i) oac[nt][i] *= corr[i];

    asm volatile("s_wait_dscnt 0x0" ::: "memory");  // P store -> frag reload
    v16h pa = ldfragA(&Pw[n0 * 32 + kk0]);
#pragma unroll
    for (int nt = 0; nt < 4; ++nt) {
      v16h bf = *(const v16h*)&Vt[(nt * 16 + n0) * 32 + kk0];
      oac[nt] = wmma16(pa, bf, oac[nt]);
    }
  }

  _Float16* Op = O + (size_t)b * LSEQ * QKVW + hd * HDIM;
#pragma unroll
  for (int i = 0; i < 8; ++i) {
    float inv = 1.0f / lsum[i];
    const int qrow = qbase + roff + i;
#pragma unroll
    for (int nt = 0; nt < 4; ++nt)
      Op[(size_t)qrow * QKVW + nt * 16 + n0] = (_Float16)(oac[nt][i] * inv);
  }
}

// ---------------------------------------------------------------------------
// Head: out[t] = sum_c Y[t,c]*W3[c] + b3  (one wave per token)
__global__ void k_head_out(const _Float16* __restrict__ Y, const float* __restrict__ W3,
                           const float* __restrict__ b3, float* __restrict__ out) {
  int lane = threadIdx.x & 31;
  int t = blockIdx.x * 8 + (threadIdx.x >> 5);
  float s = 0.f;
#pragma unroll
  for (int c = 0; c < 4; ++c)
    s += (float)Y[(size_t)t * DMODEL + lane + c * 32] * W3[lane + c * 32];
  s += __shfl_xor(s, 1, 32);  s += __shfl_xor(s, 2, 32);
  s += __shfl_xor(s, 4, 32);  s += __shfl_xor(s, 8, 32);
  s += __shfl_xor(s, 16, 32);
  if (lane == 0) out[t] = s + b3[0];
}

// ---------------------------------------------------------------------------
extern "C" void kernel_launch(void* const* d_in, const int* in_sizes, int n_in,
                              void* d_out, int out_size, void* d_ws, size_t ws_size,
                              hipStream_t stream) {
  (void)in_sizes; (void)n_in; (void)out_size; (void)ws_size;
  const float* z      = (const float*)d_in[0];
  const float* ls     = (const float*)d_in[1];
  const float* sp     = (const float*)d_in[2];
  const float* emb    = (const float*)d_in[3];
  const float* in_W1  = (const float*)d_in[4];
  const float* in_b1  = (const float*)d_in[5];
  const float* in_W2  = (const float*)d_in[6];
  const float* in_b2  = (const float*)d_in[7];
  const float* rbf_W  = (const float*)d_in[8];
  const float* rbf_b  = (const float*)d_in[9];
  const float* out_W1 = (const float*)d_in[10];
  const float* out_b1 = (const float*)d_in[11];
  const float* out_W2 = (const float*)d_in[12];
  const float* out_b2 = (const float*)d_in[13];
  const float* out_W3 = (const float*)d_in[14];
  const float* out_b3 = (const float*)d_in[15];

  const int M = BATCH * LSEQ;  // 16384 tokens

  char* base = (char*)d_ws;
  size_t off = 0;
  auto alloc = [&](size_t bytes) -> void* {
    void* p = base + off;
    off = (off + bytes + 255) & ~(size_t)255;
    return p;
  };
  auto allocH = [&](size_t n) { return (_Float16*)alloc(n * sizeof(_Float16)); };

  _Float16* Xin  = allocH((size_t)M * 256);
  _Float16* H1   = allocH((size_t)M * 512);
  float*    Xf   = (float*)alloc((size_t)M * 128 * sizeof(float));
  _Float16* Hn   = allocH((size_t)M * 128);
  _Float16* Qb   = allocH((size_t)M * 256);
  _Float16* Kb   = allocH((size_t)M * 256);
  _Float16* Vb   = allocH((size_t)M * 256);
  _Float16* Ob   = allocH((size_t)M * 256);
  float*    Bias = (float*)alloc((size_t)4 * 1024 * 1024 * sizeof(float));

  auto cvt = [&](const float* src, _Float16* dst, int n) {
    k_f32_to_f16<<<(n + 255) / 256, 256, 0, stream>>>(src, dst, n);
  };

  _Float16* in_W1h  = allocH(256 * 512); cvt(in_W1,  in_W1h,  256 * 512);
  _Float16* in_W2h  = allocH(512 * 128); cvt(in_W2,  in_W2h,  512 * 128);
  _Float16* out_W1h = allocH(128 * 512); cvt(out_W1, out_W1h, 128 * 512);
  _Float16* out_W2h = allocH(512 * 128); cvt(out_W2, out_W2h, 512 * 128);

  _Float16 *Wqh[4], *Wkh[4], *Wvh[4], *Woh[4], *Wf1h[4], *Wf2h[4];
  for (int blk = 0; blk < 4; ++blk) {
    int bi = 16 + 16 * blk;
    Wqh[blk]  = allocH(128 * 256); cvt((const float*)d_in[bi + 2],  Wqh[blk],  128 * 256);
    Wkh[blk]  = allocH(128 * 256); cvt((const float*)d_in[bi + 4],  Wkh[blk],  128 * 256);
    Wvh[blk]  = allocH(128 * 256); cvt((const float*)d_in[bi + 6],  Wvh[blk],  128 * 256);
    Woh[blk]  = allocH(256 * 128); cvt((const float*)d_in[bi + 8],  Woh[blk],  256 * 128);
    Wf1h[blk] = allocH(128 * 512); cvt((const float*)d_in[bi + 12], Wf1h[blk], 128 * 512);
    Wf2h[blk] = allocH(512 * 128); cvt((const float*)d_in[bi + 14], Wf2h[blk], 512 * 128);
  }

  auto gemm = [&](const _Float16* A, const _Float16* B, const float* bias,
                  const float* res, float* oF, _Float16* oH, int N, int K, bool gel) {
    dim3 g(N / 128, M / 64);
    if (gel) k_gemm<true ><<<g, 256, 0, stream>>>(A, B, bias, res, oF, oH, M, N, K);
    else     k_gemm<false><<<g, 256, 0, stream>>>(A, B, bias, res, oF, oH, M, N, K);
  };

  // ---- input featurization + MLP ----
  k_featurize<<<M, 256, 0, stream>>>(emb, z, ls, Xin);
  gemm(Xin, in_W1h, in_b1, nullptr, nullptr, H1, 512, 256, true);
  gemm(H1,  in_W2h, in_b2, nullptr, Xf,      nullptr, 128, 512, false);

  // ---- RBF distance bias (H,L,L) ----
  k_rbf<<<(1024 * 1024) / 256, 256, 0, stream>>>(sp, rbf_W, rbf_b, Bias);

  // ---- transformer blocks ----
  for (int blk = 0; blk < 4; ++blk) {
    int bi = 16 + 16 * blk;
    k_layernorm<<<M, 128, 0, stream>>>(Xf, (const float*)d_in[bi + 0],
                                       (const float*)d_in[bi + 1], Hn);
    gemm(Hn, Wqh[blk], (const float*)d_in[bi + 3], nullptr, nullptr, Qb, 256, 128, false);
    gemm(Hn, Wkh[blk], (const float*)d_in[bi + 5], nullptr, nullptr, Kb, 256, 128, false);
    gemm(Hn, Wvh[blk], (const float*)d_in[bi + 7], nullptr, nullptr, Vb, 256, 128, false);
    k_attn<<<BATCH * 4 * (LSEQ / 128), 256, 0, stream>>>(Qb, Kb, Vb, Bias, Ob);
    gemm(Ob, Woh[blk], (const float*)d_in[bi + 9], Xf, Xf, nullptr, 128, 256, false);
    k_layernorm<<<M, 128, 0, stream>>>(Xf, (const float*)d_in[bi + 10],
                                       (const float*)d_in[bi + 11], Hn);
    gemm(Hn, Wf1h[blk], (const float*)d_in[bi + 13], nullptr, nullptr, H1, 512, 128, true);
    gemm(H1, Wf2h[blk], (const float*)d_in[bi + 15], Xf, Xf, nullptr, 128, 512, false);
  }

  // ---- output head ----
  k_f32_to_f16<<<(M * 128 + 255) / 256, 256, 0, stream>>>(Xf, Hn, M * 128);
  gemm(Hn, out_W1h, out_b1, nullptr, nullptr, H1, 512, 128, true);
  gemm(H1, out_W2h, out_b2, nullptr, nullptr, Hn, 128, 512, true);
  k_head_out<<<M / 8, 256, 0, stream>>>(Hn, out_W3, out_b3, (float*)d_out);
}
